// Model_24034636988682
// MI455X (gfx1250) — matrix operand order repaired and checked by reading, not compiled, not run on hardware
//
#include <hip/hip_runtime.h>

// ---------------- problem constants ----------------
#define BB   64          // batch
#define TT   4096        // time steps
#define HH   15          // hidden
#define WIN  32          // time-steps per TDM window
#define NWIN (TT / WIN)

// ---------------- vector types ----------------
typedef _Float16     v16h  __attribute__((ext_vector_type(16)));
typedef _Float16     v8h   __attribute__((ext_vector_type(8)));
typedef float        v8f   __attribute__((ext_vector_type(8)));
typedef unsigned int u32x4 __attribute__((ext_vector_type(4)));
typedef int          i32x8 __attribute__((ext_vector_type(8)));
typedef int          i32x4 __attribute__((ext_vector_type(4)));

// ---------------- workspace layout (bytes) ----------------
// X    : pre-gathered, f16, [dir][b][t][16]    -> 2*B*T*16*2 = 16 MB
// Hbuf : hidden states, f16, [dir][b][t][16]   -> 16 MB
// WPK  : packed recurrence weights (B-frag layout) 2*4*32*16 halves
// WLPK : packed Wlin (B-frag layout) 3*32*16 halves
// BC   : combined gate bias, padded, 2*64 f32
// BL   : blin padded to 48 f32
#define XBYTES   ((size_t)2 * BB * TT * 16 * 2)
#define WPK_HALF (2 * 4 * 32 * 16)
#define WLPK_HALF (3 * 32 * 16)

// ---------------- fast transcendentals (critical path of recurrence) ----
__device__ __forceinline__ float fast_rcp(float x) {
#if __has_builtin(__builtin_amdgcn_rcpf)
  return __builtin_amdgcn_rcpf(x);            // raw v_rcp_f32, no Newton refine
#else
  return 1.0f / x;
#endif
}
__device__ __forceinline__ float fast_tanh(float x) {
#if __has_builtin(__builtin_amdgcn_tanhf)
  return __builtin_amdgcn_tanhf(x);           // native v_tanh_f32 (TRANS32)
#elif __has_builtin(__builtin_amdgcn_tanh_f32)
  return __builtin_amdgcn_tanh_f32(x);
#else
  return 2.0f * fast_rcp(1.0f + __expf(-2.0f * x)) - 1.0f;
#endif
}
__device__ __forceinline__ float sigm_f(float x) {
  // sigmoid(x) = 0.5*tanh(x/2) + 0.5  -> mul + tanh + fma
  return fmaf(0.5f, fast_tanh(0.5f * x), 0.5f);
}

// =====================================================================
// Kernel 1: pack weights/biases into WMMA B-fragment friendly layout.
// B-frag element j of lane l:  col n = l&15,  K = j<8 ? 8*(l>>4)+j : 16+8*(l>>4)+(j-8)
// Recurrence K layout: K 0..14 = Wih row, K 16..30 = Whh row (15 pad slots).
// Gate tiles: 0=i, 1=f, 2=g, 3=o (each padded 15 -> 16).
// =====================================================================
__global__ void pack_weights(const float* __restrict__ Wih_f, const float* __restrict__ Whh_f,
                             const float* __restrict__ bih_f, const float* __restrict__ bhh_f,
                             const float* __restrict__ Wih_b, const float* __restrict__ Whh_b,
                             const float* __restrict__ bih_b, const float* __restrict__ bhh_b,
                             const float* __restrict__ Wlin,  const float* __restrict__ blin,
                             _Float16* __restrict__ wpk, _Float16* __restrict__ wlpk,
                             float* __restrict__ bc, float* __restrict__ bl) {
  int tid = threadIdx.x;
  for (int e = tid; e < WPK_HALF; e += 256) {
    int j = e & 15, l = (e >> 4) & 31, g = (e >> 9) & 3, dir = e >> 11;
    int n = l & 15, hi = l >> 4;
    int K = (j < 8) ? (hi * 8 + j) : (16 + hi * 8 + (j - 8));
    const float* Wih = dir ? Wih_b : Wih_f;
    const float* Whh = dir ? Whh_b : Whh_f;
    float v = 0.f;
    if (n < HH) {
      int row = g * HH + n;
      if (K < 16) { if (K < HH)       v = Wih[row * HH + K]; }
      else        { int k2 = K - 16; if (k2 < HH) v = Whh[row * HH + k2]; }
    }
    wpk[e] = (_Float16)v;
  }
  for (int e = tid; e < WLPK_HALF; e += 256) {
    int j = e & 15, l = (e >> 4) & 31, ot = e >> 9;
    int n = l & 15, hi = l >> 4;
    int K = (j < 8) ? (hi * 8 + j) : (16 + hi * 8 + (j - 8));
    int o = ot * 16 + n;
    float v = 0.f;
    if (o < 45) {
      if (K < 16) { if (K < HH)       v = Wlin[o * 30 + K]; }            // hf coeffs
      else        { int k2 = K - 16; if (k2 < HH) v = Wlin[o * 30 + HH + k2]; } // hb coeffs
    }
    wlpk[e] = (_Float16)v;
  }
  for (int e = tid; e < 128; e += 256) {
    int n = e & 15, g = (e >> 4) & 3, dir = e >> 6;
    const float* bi = dir ? bih_b : bih_f;
    const float* bh = dir ? bhh_b : bhh_f;
    bc[e] = (n < HH) ? (bi[g * HH + n] + bh[g * HH + n]) : 0.f;
  }
  for (int e = tid; e < 48; e += 256) bl[e] = (e < 45) ? blin[e] : 0.f;
}

// =====================================================================
// Kernel 2: embedding gather -> f16, padded to 16 lanes, backward
// direction pre-reversed per sequence length (masked to zero past len).
// =====================================================================
__global__ void pack_x(const int* __restrict__ data, const int* __restrict__ lengths,
                       const float* __restrict__ E, _Float16* __restrict__ X) {
  int tid = blockIdx.x * blockDim.x + threadIdx.x;   // B*T threads
  int b = tid >> 12, t = tid & (TT - 1);
  int len = lengths[b];
  // forward
  int idx = data[b * TT + t];
  const float* e0 = E + (long long)idx * HH;
  _Float16* xf = X + ((long long)b * TT + t) * 16;
#pragma unroll
  for (int k = 0; k < 16; k++) xf[k] = (k < HH) ? (_Float16)e0[k] : (_Float16)0.f;
  // backward (time-reversed within valid region, zero past length)
  bool valid = t < len;
  int ts = valid ? (len - 1 - t) : 0;
  int idx2 = data[b * TT + ts];
  const float* e1 = E + (long long)idx2 * HH;
  _Float16* xb = X + ((long long)(BB + b) * TT + t) * 16;
#pragma unroll
  for (int k = 0; k < 16; k++) xb[k] = (valid && k < HH) ? (_Float16)e1[k] : (_Float16)0.f;
}

// =====================================================================
// TDM: async 2D tile load global->LDS (D# built per cdna5 CDNA5 ISA)
// tile: dim0 = tile0 elements (contiguous t,k for one batch row),
//       dim1 = 16 batch rows, row stride = stride0 elements, 2-byte elems.
// 6-arg builtin (clang-23 / therock-10.0 flavor).
// =====================================================================
__device__ __forceinline__ void tdm_load_2d(unsigned lds_off, const void* gptr,
                                            unsigned tile0, unsigned stride0) {
  unsigned long long ga = (unsigned long long)gptr;
  u32x4 g0;
  g0[0] = 1u;                                            // count=1, user mode
  g0[1] = lds_off;                                       // LDS byte address
  g0[2] = (unsigned)ga;                                  // global_addr[31:0]
  g0[3] = (unsigned)((ga >> 32) & 0x1FFFFFFu) | (2u << 30); // addr[56:32] | type=2
  i32x8 g1;
  unsigned td0 = stride0;                                // tensor_dim0 >= tile extent
  g1[0] = (int)(1u << 16);                               // data_size = 2 bytes
  g1[1] = (int)((td0 & 0xFFFFu) << 16);                  // tensor_dim0[15:0]
  g1[2] = (int)((td0 >> 16) | (16u << 16));              // tensor_dim0[31:16] | tensor_dim1=16
  g1[3] = (int)(tile0 << 16);                            // tile_dim0
  g1[4] = 16;                                            // tile_dim1=16, tile_dim2=0
  g1[5] = (int)stride0;                                  // tensor_dim0_stride[31:0]
  g1[6] = 0;                                             // stride hi / dim1_stride lo
  g1[7] = 0;
  i32x4 gz4 = {0, 0, 0, 0};
  i32x8 gz8 = {0, 0, 0, 0, 0, 0, 0, 0};
  __builtin_amdgcn_tensor_load_to_lds(g0, g1, gz4, gz4, gz8, 0);
}

// =====================================================================
// Kernel 3: the recurrence. 8 workgroups x 1 wave32.
// wave = (dir, 16-batch tile). Per step: 4x v_wmma_f32_16x16x32_f16
// (gates i,f,g,o, K=[x(15)|h(15)] padded to 32), cell update in D layout,
// h transposed back through LDS for the next step's A fragment.
// X windows streamed into LDS by the TDM, double buffered.
// =====================================================================
__global__ void __launch_bounds__(32)
lstm_recurrence(const _Float16* __restrict__ X, _Float16* __restrict__ Hout,
                const _Float16* __restrict__ wpk, const float* __restrict__ bc,
                const float* __restrict__ h0, const float* __restrict__ c0) {
  __shared__ _Float16 xwin[2][WIN * 16 * 16];  // [buf][batch][t_in_win][16]
  __shared__ _Float16 hbuf[16 * 16];           // [batch][unit]

  int lane = threadIdx.x;
  int dir  = blockIdx.x >> 2;
  int b0   = (blockIdx.x & 3) * 16;
  int n    = lane & 15;          // D/C column (gate/hidden unit)
  int hi   = lane >> 4;
  int mA   = lane & 15;          // A row (batch)

  // B fragments (weights) + bias-as-C fragments
  v16h w[4]; v8f cb[4];
#pragma unroll
  for (int g = 0; g < 4; g++) {
    w[g] = *(const v16h*)(wpk + ((dir * 4 + g) * 32 + lane) * 16);
    float bv = bc[dir * 64 + g * 16 + n];
#pragma unroll
    for (int r = 0; r < 8; r++) cb[g][r] = bv;
  }

  // c state lives in registers in D layout; h0 seeds the LDS transpose buffer
  v8f c;
#pragma unroll
  for (int r = 0; r < 8; r++) {
    int m = r + hi * 8;
    c[r] = (n < HH) ? c0[(dir * BB + b0 + m) * HH + n] : 0.f;
  }
  for (int e = lane; e < 256; e += 32) {
    int m = e >> 4, k = e & 15;
    hbuf[e] = (k < HH) ? (_Float16)h0[(dir * BB + b0 + m) * HH + k] : (_Float16)0.f;
  }

  const _Float16* xbase = X + (long long)(dir * BB + b0) * TT * 16;
  tdm_load_2d((unsigned)(unsigned long long)&xwin[0][0], xbase, WIN * 16, TT * 16);

  for (int wd = 0; wd < NWIN; wd++) {
    int buf = wd & 1;
    __builtin_amdgcn_s_wait_tensorcnt(0);          // window wd resident
    if (wd + 1 < NWIN)                              // prefetch next window
      tdm_load_2d((unsigned)(unsigned long long)&xwin[buf ^ 1][0],
                  xbase + (long long)(wd + 1) * WIN * 16, WIN * 16, TT * 16);

    for (int tw = 0; tw < WIN; tw++) {
      int t = wd * WIN + tw;
      // A fragment: elems 0..7 = x (K=(hi?8:0)+0..7), 8..15 = h (K=16+...)
      v8h xh = *(const v8h*)&xwin[buf][(mA * WIN + tw) * 16 + hi * 8];
      v8h hh = *(const v8h*)&hbuf[mA * 16 + hi * 8];
      v16h a;
#pragma unroll
      for (int k = 0; k < 8; k++) { a[k] = xh[k]; a[8 + k] = hh[k]; }

      v8f gi = __builtin_amdgcn_wmma_f32_16x16x32_f16(false, a, false, w[0], (short)0, cb[0], false, false);
      v8f gf = __builtin_amdgcn_wmma_f32_16x16x32_f16(false, a, false, w[1], (short)0, cb[1], false, false);
      v8f gg = __builtin_amdgcn_wmma_f32_16x16x32_f16(false, a, false, w[2], (short)0, cb[2], false, false);
      v8f go = __builtin_amdgcn_wmma_f32_16x16x32_f16(false, a, false, w[3], (short)0, cb[3], false, false);

#pragma unroll
      for (int r = 0; r < 8; r++) {
        float iv = sigm_f(gi[r]);
        float fv = sigm_f(gf[r]);
        float gv = fast_tanh(gg[r]);
        float ov = sigm_f(go[r]);
        float cv = fv * c[r] + iv * gv;
        c[r] = cv;
        _Float16 hv = (_Float16)(ov * fast_tanh(cv));
        int m = r + hi * 8;
        hbuf[m * 16 + n] = hv;                                          // transpose for next step
        Hout[((long long)(dir * BB + b0 + m) * TT + t) * 16 + n] = hv;  // persist for output stage
      }
    }
  }
}

// =====================================================================
// Kernel 4: out = [hf | hb_unreversed] @ Wlin^T + blin, masked past length
// (invalid rows -> zero A rows -> out = blin, matching the reference).
// One wave per 16-timestep tile, 3 WMMAs per tile (45 -> 3x16 outputs).
// =====================================================================
__global__ void __launch_bounds__(256)
output_linear(const _Float16* __restrict__ Hbuf, const int* __restrict__ lengths,
              const _Float16* __restrict__ wlpk, const float* __restrict__ bl,
              float* __restrict__ out) {
  int lane = threadIdx.x & 31;
  int wv   = blockIdx.x * (blockDim.x >> 5) + (threadIdx.x >> 5);
  int b    = wv >> 8;               // T/16 = 256 tiles per batch row
  int tt0  = (wv & 255) * 16;
  int n = lane & 15, hi = lane >> 4, mA = lane & 15;
  int len = lengths[b];

  v16h wl[3]; v8f cbl[3];
#pragma unroll
  for (int ot = 0; ot < 3; ot++) {
    wl[ot] = *(const v16h*)(wlpk + (ot * 32 + lane) * 16);
    float bv = bl[ot * 16 + n];
#pragma unroll
    for (int r = 0; r < 8; r++) cbl[ot][r] = bv;
  }

  int tA = tt0 + mA;
  v8h hf, hb;
#pragma unroll
  for (int k = 0; k < 8; k++) { hf[k] = (_Float16)0.f; hb[k] = (_Float16)0.f; }
  if (tA < len) {
    hf = *(const v8h*)(Hbuf + ((long long)b * TT + tA) * 16 + hi * 8);
    int trev = len - 1 - tA;  // un-reverse the backward direction
    hb = *(const v8h*)(Hbuf + ((long long)(BB + b) * TT + trev) * 16 + hi * 8);
  }
  v16h a;
#pragma unroll
  for (int k = 0; k < 8; k++) { a[k] = hf[k]; a[8 + k] = hb[k]; }

#pragma unroll
  for (int ot = 0; ot < 3; ot++) {
    v8f d = __builtin_amdgcn_wmma_f32_16x16x32_f16(false, a, false, wl[ot], (short)0, cbl[ot], false, false);
    int o = ot * 16 + n;
    if (o < 45) {
#pragma unroll
      for (int r = 0; r < 8; r++)
        out[((long long)b * TT + tt0 + r + hi * 8) * 45 + o] = d[r];
    }
  }
}

// =====================================================================
extern "C" void kernel_launch(void* const* d_in, const int* in_sizes, int n_in,
                              void* d_out, int out_size, void* d_ws, size_t ws_size,
                              hipStream_t stream) {
  const int*   data    = (const int*)d_in[0];
  const int*   lengths = (const int*)d_in[2];
  const float* E       = (const float*)d_in[3];
  const float* Wih_f   = (const float*)d_in[4];
  const float* Whh_f   = (const float*)d_in[5];
  const float* bih_f   = (const float*)d_in[6];
  const float* bhh_f   = (const float*)d_in[7];
  const float* Wih_b   = (const float*)d_in[8];
  const float* Whh_b   = (const float*)d_in[9];
  const float* bih_b   = (const float*)d_in[10];
  const float* bhh_b   = (const float*)d_in[11];
  const float* Wlin    = (const float*)d_in[12];
  const float* blin    = (const float*)d_in[13];
  const float* h0      = (const float*)d_in[14];
  const float* c0      = (const float*)d_in[15];

  char* ws = (char*)d_ws;
  _Float16* X    = (_Float16*)(ws);
  _Float16* Hbuf = (_Float16*)(ws + XBYTES);
  _Float16* wpk  = (_Float16*)(ws + 2 * XBYTES);
  _Float16* wlpk = wpk + WPK_HALF;
  float*    bc   = (float*)(wlpk + WLPK_HALF);
  float*    bl   = bc + 128;

  pack_weights<<<1, 256, 0, stream>>>(Wih_f, Whh_f, bih_f, bhh_f,
                                      Wih_b, Whh_b, bih_b, bhh_b,
                                      Wlin, blin, wpk, wlpk, bc, bl);
  pack_x<<<(BB * TT) / 256, 256, 0, stream>>>(data, lengths, E, X);
  lstm_recurrence<<<8, 32, 0, stream>>>(X, Hbuf, wpk, bc, h0, c0);
  output_linear<<<(BB * (TT / 16)) / 8, 256, 0, stream>>>(Hbuf, lengths, wlpk, bl, (float*)d_out);
}
